// GDTA_block_47588237640036
// MI455X (gfx1250) — compile-verified, matrix operand order checked by reading, not run on hardware
//
#include <hip/hip_runtime.h>
#include <hip/hip_bf16.h>
#include <stdint.h>

// ---------------- constants ----------------
#define BATCH   16
#define HH      56
#define WW      56
#define DIM     256
#define NPIX    (HH*WW)              // 3136
#define ROWS    (BATCH*NPIX)         // 50176
#define HEADS   8
#define HD      32
#define GWC     64                   // DIM/GROUPS
#define HIDDEN  1024
#define SCALE_F 0.17677669529663687f // 32^-0.5

typedef __bf16 bf16_t;
typedef __attribute__((ext_vector_type(16))) __bf16 v16bf;
typedef __attribute__((ext_vector_type(8)))  __bf16 v8bf;
typedef __attribute__((ext_vector_type(8)))  float  v8f;
typedef __attribute__((ext_vector_type(4)))  int    v4i;

#if defined(__AMDGCN__) && __has_builtin(__builtin_amdgcn_global_load_async_to_lds_b128)
#define HAVE_ASYNC 1
#else
#define HAVE_ASYNC 0
#endif

#if HAVE_ASYNC
// 16-byte global -> LDS async copy (ASYNCcnt-tracked, no staging VGPRs)
static __device__ inline void async_cp16(void* lds, const void* g) {
    __attribute__((address_space(3))) v4i* l =
        (__attribute__((address_space(3))) v4i*)(unsigned int)(uintptr_t)lds;
    __attribute__((address_space(1))) v4i* gp =
        (__attribute__((address_space(1))) v4i*)(uintptr_t)g;
    __builtin_amdgcn_global_load_async_to_lds_b128(gp, l, 0, 0);
}
// wait until at most N async ops remain outstanding (in-order completion =>
// everything older than the newest N has landed)
template <int N>
static __device__ inline void async_waitN() {
#if __has_builtin(__builtin_amdgcn_s_wait_asynccnt)
    __builtin_amdgcn_s_wait_asynccnt(N);
#else
    asm volatile("s_wait_asynccnt %0" :: "i"(N) : "memory");
#endif
}
#endif

static __device__ inline float bf2f(unsigned short u) {
    unsigned int x = ((unsigned int)u) << 16;
    float f; __builtin_memcpy(&f, &x, 4); return f;
}
static __device__ inline unsigned short f2bf(float f) {
    __bf16 h = (__bf16)f;
    unsigned short u; __builtin_memcpy(&u, &h, 2); return u;
}

// 16x32 bf16 fragment from LDS per CDNA5 16-bit A layout:
// lane<16: row=lane, k in {0..7,16..23}; lane>=16: row=lane-16, k in {8..15,24..31}
static __device__ inline v16bf ld_frag(const bf16_t* base, int rbase, int pitch,
                                       int kbase, int lane) {
    int r  = rbase + (lane & 15);
    int ko = (lane & 16) ? 8 : 0;
    const bf16_t* p = base + (size_t)r * pitch + kbase + ko;
    v8bf a = *(const v8bf*)p;
    v8bf b = *(const v8bf*)(p + 16);
    return __builtin_shufflevector(a, b, 0,1,2,3,4,5,6,7,8,9,10,11,12,13,14,15);
}

static __device__ inline v8f wmma_bf16(v16bf a, v16bf b, v8f c) {
    return __builtin_amdgcn_wmma_f32_16x16x32_bf16(false, a, false, b,
                                                   (short)0, c, false, false);
}

// ---------------- 0a. weight transpose + fp32->bf16 ----------------
__global__ void k_tconv(const float* __restrict__ src, unsigned short* __restrict__ dst,
                        int K, int N) {
    int idx = blockIdx.x * 256 + threadIdx.x;
    if (idx < K * N) {
        int k = idx / N, n = idx % N;
        dst[(size_t)n * K + k] = f2bf(src[idx]);
    }
}

// ---------------- 0b. BN affine precompute ----------------
__global__ void k_bnaff(const float* g, const float* bt, const float* mean,
                        const float* var, float* affa, float* affb) {
    int c = threadIdx.x;
    float a = g[c] * rsqrtf(var[c] + 1e-3f);
    affa[c] = a;
    affb[c] = bt[c] - mean[c] * a;
}

// ---------------- 1. dwconv + group cumsum + LayerNorm ----------------
__global__ void k_dwln(const float* __restrict__ x, const float* __restrict__ kern,
                       const float* __restrict__ kbias, const float* __restrict__ lng,
                       const float* __restrict__ lnb, float* __restrict__ dwc,
                       unsigned short* __restrict__ xn) {
    int pix = blockIdx.x;
    int c = threadIdx.x;
    int w = pix % WW, h = (pix / WW) % HH, b = pix / (WW * HH);
    __shared__ float kbuf[192];
    __shared__ float red[256];

    float val = 0.f;
    if (c < GWC) {
        val = x[(size_t)pix * DIM + c];
    } else {
        int cc = c - GWC;
        float s = kbias[cc];
#pragma unroll
        for (int dh = 0; dh < 3; dh++) {
#pragma unroll
            for (int dw = 0; dw < 3; dw++) {
                int ih = h + dh - 1, iw = w + dw - 1;
                if (ih >= 0 && ih < HH && iw >= 0 && iw < WW) {
                    s += x[(((size_t)b * HH + ih) * WW + iw) * DIM + c] *
                         kern[(dh * 3 + dw) * 192 + cc];
                }
            }
        }
        kbuf[cc] = s;
    }
    __syncthreads();
    if (c >= GWC) {
        int cc = c - GWC, g = cc >> 6, l = cc & 63;
        float s = kbuf[l];
        if (g >= 1) s += kbuf[64 + l];
        if (g >= 2) s += kbuf[128 + l];
        val = s;
    }
    red[c] = val; __syncthreads();
    for (int s = 128; s > 0; s >>= 1) { if (c < s) red[c] += red[c + s]; __syncthreads(); }
    float mu = red[0] * (1.f / DIM);
    __syncthreads();
    float d = val - mu;
    red[c] = d * d; __syncthreads();
    for (int s = 128; s > 0; s >>= 1) { if (c < s) red[c] += red[c + s]; __syncthreads(); }
    float var = red[0] * (1.f / DIM);
    float nv = d * rsqrtf(var + 1e-6f) * lng[c] + lnb[c];
    dwc[(size_t)pix * DIM + c] = val;
    xn[(size_t)pix * DIM + c] = f2bf(nv);
}

// ---------------- 2. WMMA GEMM: C = A(MxK) . BT(NxK)^T ----------------
// block 64x256, 8 waves of 32x64, triple-buffered async LDS staging.
template <int EPI>
__global__ void k_gemm(const unsigned short* __restrict__ A,
                       const unsigned short* __restrict__ BT,
                       unsigned short* __restrict__ Co,
                       const float* __restrict__ bias,
                       const float* __restrict__ resid,
                       const float* __restrict__ affa,
                       const float* __restrict__ affb,
                       int M, int N, int K) {
    __shared__ __align__(16) bf16_t Al[3][64 * 40];
    __shared__ __align__(16) bf16_t Bl[3][256 * 40];
    int t = threadIdx.x, wave = t >> 5, lane = t & 31;
    int wm = wave >> 2, wn = wave & 3;
    int row0 = blockIdx.x * 64, col0 = blockIdx.y * 256;

    int sm = t >> 2, sp = t & 3;
    const unsigned short* Ag = A + (size_t)(row0 + sm) * K + sp * 8;
    const unsigned short* Bg = BT + (size_t)(col0 + t) * K;
    bf16_t* Ald = &Al[0][sm * 40 + sp * 8];
    bf16_t* Bld = &Bl[0][t * 40];

    auto stage = [&](int buf, int k0) {
        bf16_t* ad = Ald + buf * (64 * 40);
        bf16_t* bd = Bld + buf * (256 * 40);
#if HAVE_ASYNC
        async_cp16(ad, Ag + k0);
        async_cp16(bd,      Bg + k0);
        async_cp16(bd + 8,  Bg + k0 + 8);
        async_cp16(bd + 16, Bg + k0 + 16);
        async_cp16(bd + 24, Bg + k0 + 24);
#else
        uint4 ra  = *(const uint4*)(Ag + k0);
        uint4 rb0 = *(const uint4*)(Bg + k0);
        uint4 rb1 = *(const uint4*)(Bg + k0 + 8);
        uint4 rb2 = *(const uint4*)(Bg + k0 + 16);
        uint4 rb3 = *(const uint4*)(Bg + k0 + 24);
        *(uint4*)ad = ra;
        *(uint4*)(bd)      = rb0;
        *(uint4*)(bd + 8)  = rb1;
        *(uint4*)(bd + 16) = rb2;
        *(uint4*)(bd + 24) = rb3;
#endif
    };

    v8f zero = {0,0,0,0,0,0,0,0};
    v8f acc[2][4];
#pragma unroll
    for (int i = 0; i < 2; i++)
#pragma unroll
        for (int j = 0; j < 4; j++) acc[i][j] = zero;

    stage(0, 0);
    int nk = K >> 5;
    for (int ks = 0; ks < nk; ks++) {
        // issue next prefetch FIRST, then wait with threshold = ops/stage so the
        // new batch stays in flight while the needed (older) batch is forced done.
        if (ks + 1 < nk) {
            stage((ks + 1) % 3, (ks + 1) * 32);
#if HAVE_ASYNC
            async_waitN<5>();
#endif
        } else {
#if HAVE_ASYNC
            async_waitN<0>();
#endif
        }
        __syncthreads();
        const bf16_t* Ac = Al[ks % 3];
        const bf16_t* Bc = Bl[ks % 3];
        v16bf a0 = ld_frag(Ac, wm * 32,      40, 0, lane);
        v16bf a1 = ld_frag(Ac, wm * 32 + 16, 40, 0, lane);
        v16bf b0 = ld_frag(Bc, wn * 64,      40, 0, lane);
        v16bf b1 = ld_frag(Bc, wn * 64 + 16, 40, 0, lane);
        v16bf b2 = ld_frag(Bc, wn * 64 + 32, 40, 0, lane);
        v16bf b3 = ld_frag(Bc, wn * 64 + 48, 40, 0, lane);
        acc[0][0] = wmma_bf16(a0, b0, acc[0][0]);
        acc[0][1] = wmma_bf16(a0, b1, acc[0][1]);
        acc[0][2] = wmma_bf16(a0, b2, acc[0][2]);
        acc[0][3] = wmma_bf16(a0, b3, acc[0][3]);
        acc[1][0] = wmma_bf16(a1, b0, acc[1][0]);
        acc[1][1] = wmma_bf16(a1, b1, acc[1][1]);
        acc[1][2] = wmma_bf16(a1, b2, acc[1][2]);
        acc[1][3] = wmma_bf16(a1, b3, acc[1][3]);
    }
    int lm = lane & 15, hi = lane >> 4;
#pragma unroll
    for (int i = 0; i < 2; i++)
#pragma unroll
        for (int j = 0; j < 4; j++)
#pragma unroll
            for (int r = 0; r < 8; r++) {
                int m = row0 + wm * 32 + i * 16 + r + hi * 8;
                int n = col0 + wn * 64 + j * 16 + lm;
                float v = acc[i][j][r];
                if (EPI == 1) {
                    v += bias[n] + resid[(size_t)m * N + n];
                    v = v * affa[n] + affb[n];
                }
                Co[(size_t)m * N + n] = f2bf(v);
            }
}

// ---------------- 3. attention Gram matrix + softmax ----------------
__global__ void k_attn(const unsigned short* __restrict__ qkv, float* __restrict__ attn) {
    int bh = blockIdx.x, b = bh >> 3, h = bh & 7;
    int t = threadIdx.x;
    __shared__ float qs[64][32];
    __shared__ float ks[64][32];
    __shared__ float am[32 * 32];
    float acc[4] = {0, 0, 0, 0};
    int d = t & 31, e0 = (t >> 5) * 4;
    int lr = t >> 2, part = t & 3;

    for (int ch = 0; ch < 49; ch++) {
        int row = b * NPIX + ch * 64 + lr;
        const unsigned short* qp = qkv + (size_t)row * 768 + h * HD + part * 8;
        uint4 rq = *(const uint4*)qp;
        const unsigned short* qu = (const unsigned short*)&rq;
        float qv[8], ssq = 0.f;
#pragma unroll
        for (int j = 0; j < 8; j++) { qv[j] = bf2f(qu[j]); ssq += qv[j] * qv[j]; }
        ssq += __shfl_xor(ssq, 1, 32); ssq += __shfl_xor(ssq, 2, 32);
        float qinv = rsqrtf(fmaxf(ssq, 1e-12f)) * SCALE_F;

        const unsigned short* kp = qkv + (size_t)row * 768 + DIM + h * HD + part * 8;
        uint4 rk = *(const uint4*)kp;
        const unsigned short* ku = (const unsigned short*)&rk;
        float kv[8], ssk = 0.f;
#pragma unroll
        for (int j = 0; j < 8; j++) { kv[j] = bf2f(ku[j]); ssk += kv[j] * kv[j]; }
        ssk += __shfl_xor(ssk, 1, 32); ssk += __shfl_xor(ssk, 2, 32);
        float kinv = rsqrtf(fmaxf(ssk, 1e-12f));

#pragma unroll
        for (int j = 0; j < 8; j++) {
            qs[lr][part * 8 + j] = qv[j] * qinv;
            ks[lr][part * 8 + j] = kv[j] * kinv;
        }
        __syncthreads();
        for (int n = 0; n < 64; n++) {
            float qd = qs[n][d];
            acc[0] += qd * ks[n][e0];
            acc[1] += qd * ks[n][e0 + 1];
            acc[2] += qd * ks[n][e0 + 2];
            acc[3] += qd * ks[n][e0 + 3];
        }
        __syncthreads();
    }
#pragma unroll
    for (int j = 0; j < 4; j++) am[d * 32 + e0 + j] = acc[j];
    __syncthreads();
    if (t < 32) {
        float m = -3.4e38f;
        for (int e = 0; e < 32; e++) m = fmaxf(m, am[t * 32 + e]);
        float s = 0.f;
        for (int e = 0; e < 32; e++) s += __expf(am[t * 32 + e] - m);
        float inv = 1.f / s;
        for (int e = 0; e < 32; e++)
            attn[(size_t)bh * 1024 + t * 32 + e] = __expf(am[t * 32 + e] - m) * inv;
    }
}

// ---------------- 4. out = v @ attn (per head), 32 rows per block ----------------
__global__ void k_attnout(const unsigned short* __restrict__ qkv,
                          const float* __restrict__ attn,
                          unsigned short* __restrict__ aout) {
    int r0 = blockIdx.x * 32;
    int b = r0 / NPIX;
    int t = threadIdx.x;
    __shared__ float at[8 * 32 * 32];
    __shared__ __align__(16) unsigned short vs[32 * 256];
    for (int i = 0; i < 32; i++)
        at[i * 256 + t] = attn[(size_t)b * 8192 + i * 256 + t];
    for (int i = 0; i < 4; i++) {
        int idx = t + i * 256, lr = idx >> 5, pc = idx & 31;
        uint4 raw = *(const uint4*)(qkv + (size_t)(r0 + lr) * 768 + 512 + pc * 8);
        *(uint4*)(vs + lr * 256 + pc * 8) = raw;
    }
    __syncthreads();
    int h = t >> 5, e = t & 31;
    for (int lr = 0; lr < 32; lr++) {
        float a = 0.f;
#pragma unroll 8
        for (int dd = 0; dd < 32; dd++)
            a += bf2f(vs[lr * 256 + h * 32 + dd]) * at[h * 1024 + dd * 32 + e];
        aout[(size_t)(r0 + lr) * DIM + t] = f2bf(a);
    }
}

// ---------------- 5. fused MLP (h1 LDS-resident) ----------------
// 32 rows/block; dyn LDS: Asm 32x264, H1 32x1032, Bst 3x[256x40] (bf16)
__global__ void k_mlp(const unsigned short* __restrict__ Mg,
                      const unsigned short* __restrict__ fc1T,
                      const unsigned short* __restrict__ fc2T,
                      const float* __restrict__ b1, const float* __restrict__ b2,
                      const float* __restrict__ xres, float* __restrict__ out) {
    extern __shared__ __align__(16) unsigned char smem[];
    bf16_t* Asm = (bf16_t*)smem;                                  // [32][264]
    bf16_t* H1  = (bf16_t*)(smem + 32 * 264 * 2);                 // [32][1032]
    bf16_t* Bst = (bf16_t*)(smem + 32 * 264 * 2 + 32 * 1032 * 2); // 3 x [256][40]

    int t = threadIdx.x, wave = t >> 5, lane = t & 31;
    int wm = wave >> 2, wn = wave & 3;
    int row0 = blockIdx.x * 32;
    int lm = lane & 15, hi = lane >> 4;
    v8f zero = {0,0,0,0,0,0,0,0};

    // one-time A (merged) stage 32x256
    {
#if HAVE_ASYNC
        for (int i = 0; i < 4; i++) {
            int idx = t + i * 256, m = idx >> 5, pc = idx & 31;
            async_cp16(Asm + m * 264 + pc * 8, Mg + (size_t)(row0 + m) * DIM + pc * 8);
        }
#else
        for (int i = 0; i < 4; i++) {
            int idx = t + i * 256, m = idx >> 5, pc = idx & 31;
            uint4 raw = *(const uint4*)(Mg + (size_t)(row0 + m) * DIM + pc * 8);
            *(uint4*)((unsigned short*)Asm + m * 264 + pc * 8) = raw;
        }
#endif
    }

    int s1n = t >> 1, s1h = t & 1;
    // GEMM1 staging for flattened step: N0=(step>>3)*128, k0=(step&7)*32
    auto stage1 = [&](int buf, int step) {
        int N0 = (step >> 3) * 128, k0 = (step & 7) * 32;
        bf16_t* bd = Bst + buf * (256 * 40) + s1n * 40 + s1h * 16;
        const unsigned short* src = fc1T + (size_t)(N0 + s1n) * 256 + k0 + s1h * 16;
#if HAVE_ASYNC
        async_cp16(bd, src);
        async_cp16(bd + 8, src + 8);
#else
        uint4 r0 = *(const uint4*)src;
        uint4 r1 = *(const uint4*)(src + 8);
        *(uint4*)bd = r0;
        *(uint4*)(bd + 8) = r1;
#endif
    };

    // ---- GEMM1: 32x256 @ 256x1024 -> H1 (+bias, gelu), 8 col tiles x 8 k steps ----
    stage1(0, 0);
    for (int nt = 0; nt < 8; nt++) {
        int N0 = nt * 128;
        v8f acc[2]; acc[0] = zero; acc[1] = zero;
        for (int ks = 0; ks < 8; ks++) {
            int step = nt * 8 + ks;
            if (step + 1 < 64) {
                stage1((step + 1) % 3, step + 1);
#if HAVE_ASYNC
                async_waitN<2>();
#endif
            } else {
#if HAVE_ASYNC
                async_waitN<0>();
#endif
            }
            __syncthreads();
            v16bf a0  = ld_frag(Asm, wm * 16, 264, ks * 32, lane);
            const bf16_t* Bc = Bst + (step % 3) * (256 * 40);
            v16bf bb0 = ld_frag(Bc, wn * 32,      40, 0, lane);
            v16bf bb1 = ld_frag(Bc, wn * 32 + 16, 40, 0, lane);
            acc[0] = wmma_bf16(a0, bb0, acc[0]);
            acc[1] = wmma_bf16(a0, bb1, acc[1]);
        }
#pragma unroll
        for (int j = 0; j < 2; j++)
#pragma unroll
            for (int r = 0; r < 8; r++) {
                int m = wm * 16 + r + hi * 8;
                int n = N0 + wn * 32 + j * 16 + lm;
                float xg = acc[j][r] + b1[n];
                float ge = 0.5f * xg * (1.f + erff(xg * 0.70710678118654752f));
                H1[m * 1032 + n] = (__bf16)ge;
            }
    }

    auto stage2 = [&](int buf, int k0) {
        bf16_t* bd = Bst + buf * (256 * 40) + t * 40;
        const unsigned short* src = fc2T + (size_t)t * 1024 + k0;
#if HAVE_ASYNC
        async_cp16(bd,      src);
        async_cp16(bd + 8,  src + 8);
        async_cp16(bd + 16, src + 16);
        async_cp16(bd + 24, src + 24);
#else
        uint4 q0 = *(const uint4*)src;
        uint4 q1 = *(const uint4*)(src + 8);
        uint4 q2 = *(const uint4*)(src + 16);
        uint4 q3 = *(const uint4*)(src + 24);
        *(uint4*)(bd)      = q0;
        *(uint4*)(bd + 8)  = q1;
        *(uint4*)(bd + 16) = q2;
        *(uint4*)(bd + 24) = q3;
#endif
    };

    // ---- GEMM2: 32x1024 @ 1024x256 -> out (+bias +x residual) ----
    stage2(0, 0);
    __syncthreads();   // H1 complete for all waves
    v8f acc2[4]; acc2[0] = zero; acc2[1] = zero; acc2[2] = zero; acc2[3] = zero;
    for (int ks = 0; ks < 32; ks++) {
        if (ks + 1 < 32) {
            stage2((ks + 1) % 3, (ks + 1) * 32);
#if HAVE_ASYNC
            async_waitN<4>();
#endif
        } else {
#if HAVE_ASYNC
            async_waitN<0>();
#endif
        }
        __syncthreads();
        v16bf a0 = ld_frag(H1, wm * 16, 1032, ks * 32, lane);
        const bf16_t* Bc = Bst + (ks % 3) * (256 * 40);
        v16bf bb0 = ld_frag(Bc, wn * 64,      40, 0, lane);
        v16bf bb1 = ld_frag(Bc, wn * 64 + 16, 40, 0, lane);
        v16bf bb2 = ld_frag(Bc, wn * 64 + 32, 40, 0, lane);
        v16bf bb3 = ld_frag(Bc, wn * 64 + 48, 40, 0, lane);
        acc2[0] = wmma_bf16(a0, bb0, acc2[0]);
        acc2[1] = wmma_bf16(a0, bb1, acc2[1]);
        acc2[2] = wmma_bf16(a0, bb2, acc2[2]);
        acc2[3] = wmma_bf16(a0, bb3, acc2[3]);
    }
#pragma unroll
    for (int j = 0; j < 4; j++)
#pragma unroll
        for (int r = 0; r < 8; r++) {
            int m = wm * 16 + r + hi * 8;
            int n = wn * 64 + j * 16 + lm;
            size_t gi = (size_t)(row0 + m) * DIM + n;
            out[gi] = acc2[j][r] + b2[n] + xres[gi];
        }
}

// ---------------- host ----------------
extern "C" void kernel_launch(void* const* d_in, const int* in_sizes, int n_in,
                              void* d_out, int out_size, void* d_ws, size_t ws_size,
                              hipStream_t stream) {
    const float* x        = (const float*)d_in[0];
    const float* dw_kern  = (const float*)d_in[1];
    const float* dw_bias  = (const float*)d_in[2];
    const float* ln_gamma = (const float*)d_in[3];
    const float* ln_beta  = (const float*)d_in[4];
    const float* qkv_w    = (const float*)d_in[5];
    const float* proj_w   = (const float*)d_in[6];
    const float* proj_b   = (const float*)d_in[7];
    const float* bn_gamma = (const float*)d_in[8];
    const float* bn_beta  = (const float*)d_in[9];
    const float* bn_mean  = (const float*)d_in[10];
    const float* bn_var   = (const float*)d_in[11];
    const float* fc1_w    = (const float*)d_in[12];
    const float* fc1_b    = (const float*)d_in[13];
    const float* fc2_w    = (const float*)d_in[14];
    const float* fc2_b    = (const float*)d_in[15];
    float* out = (float*)d_out;

    size_t off = 0;
    auto take = [&](size_t bytes) { size_t o = off; off += (bytes + 255) & ~(size_t)255; return o; };
    size_t o_qkvT  = take((size_t)768 * 256 * 2);
    size_t o_projT = take((size_t)256 * 256 * 2);
    size_t o_fc1T  = take((size_t)1024 * 256 * 2);
    size_t o_fc2T  = take((size_t)256 * 1024 * 2);
    size_t o_affa  = take(256 * 4);
    size_t o_affb  = take(256 * 4);
    size_t o_dw    = take((size_t)ROWS * DIM * 4);
    size_t o_x16   = take((size_t)ROWS * DIM * 2);   // norm1; reused for attn_out
    size_t o_qkv   = take((size_t)ROWS * 768 * 2);   // qkv; reused for merged
    size_t o_attn  = take((size_t)BATCH * HEADS * 1024 * 4);
    if (ws_size < off) return;

    unsigned char* ws = (unsigned char*)d_ws;
    unsigned short* qkvT   = (unsigned short*)(ws + o_qkvT);
    unsigned short* projT  = (unsigned short*)(ws + o_projT);
    unsigned short* fc1T   = (unsigned short*)(ws + o_fc1T);
    unsigned short* fc2T   = (unsigned short*)(ws + o_fc2T);
    float*          affa   = (float*)(ws + o_affa);
    float*          affb   = (float*)(ws + o_affb);
    float*          dwc    = (float*)(ws + o_dw);
    unsigned short* x16    = (unsigned short*)(ws + o_x16);
    unsigned short* qkvb   = (unsigned short*)(ws + o_qkv);
    float*          attn   = (float*)(ws + o_attn);
    unsigned short* aout   = x16;
    unsigned short* merged = qkvb;

    k_tconv<<<(256 * 768 + 255) / 256, 256, 0, stream>>>(qkv_w, qkvT, 256, 768);
    k_tconv<<<(256 * 256 + 255) / 256, 256, 0, stream>>>(proj_w, projT, 256, 256);
    k_tconv<<<(256 * 1024 + 255) / 256, 256, 0, stream>>>(fc1_w, fc1T, 256, 1024);
    k_tconv<<<(1024 * 256 + 255) / 256, 256, 0, stream>>>(fc2_w, fc2T, 1024, 256);
    k_bnaff<<<1, 256, 0, stream>>>(bn_gamma, bn_beta, bn_mean, bn_var, affa, affb);

    k_dwln<<<ROWS, 256, 0, stream>>>(x, dw_kern, dw_bias, ln_gamma, ln_beta, dwc, x16);

    // qkv: 50176 x 768 x 256
    k_gemm<0><<<dim3(ROWS / 64, 768 / 256), 256, 0, stream>>>(
        x16, qkvT, qkvb, nullptr, nullptr, nullptr, nullptr, ROWS, 768, 256);

    k_attn<<<BATCH * HEADS, 256, 0, stream>>>(qkvb, attn);
    k_attnout<<<ROWS / 32, 256, 0, stream>>>(qkvb, attn, aout);

    // proj + bias + dwconv residual + BN affine -> merged bf16
    k_gemm<1><<<dim3(ROWS / 64, 256 / 256), 256, 0, stream>>>(
        aout, projT, merged, proj_b, dwc, affa, affb, ROWS, 256, 256);

    size_t smem = (size_t)32 * 264 * 2 + (size_t)32 * 1032 * 2 + 3 * (size_t)256 * 40 * 2;
    k_mlp<<<ROWS / 32, 256, smem, stream>>>(merged, fc1T, fc2T, fc1_b, fc2_b, x, out);
}
